// PinSAGE_20779051778133
// MI455X (gfx1250) — matrix operand order repaired and checked by reading, not compiled, not run on hardware
//
#include <hip/hip_runtime.h>

// ---------------------------------------------------------------------------
// PinSAGE forward on MI455X (gfx1250): bf16 WMMA GEMMs + f32 everywhere else.
// Weights are pre-converted to bf16 AND pre-transposed ([n][k]) once, so the
// per-workgroup LDS staging is a straight contiguous copy -> done with
// GLOBAL_LOAD_ASYNC_TO_LDS_B128 (ASYNCcnt-tracked, bypasses VGPRs).
// ---------------------------------------------------------------------------

typedef __attribute__((ext_vector_type(16))) __bf16 v16bf;
typedef __attribute__((ext_vector_type(8)))  float  v8f;
typedef __attribute__((ext_vector_type(4)))  int    v4i;

#define DD 128
constexpr int N0 = 1048576, N1 = 131072, N2 = 16384;
constexpr int E0 = 1048576, E1 = 131072, EP = 16384;

// ---- workspace layout (bytes) ----
constexpr size_t OFF_MACC0 = 0;
constexpr size_t OFF_WSUM0 = OFF_MACC0 + (size_t)N1 * DD * 4;
constexpr size_t OFF_MACC1 = OFF_WSUM0 + (size_t)N1 * 4;
constexpr size_t OFF_WSUM1 = OFF_MACC1 + (size_t)N2 * DD * 4;
constexpr size_t ZERO_BYTES = OFF_WSUM1 + (size_t)N2 * 4;
constexpr size_t OFF_N0BF  = ZERO_BYTES;
constexpr size_t OFF_N1BF  = OFF_N0BF + (size_t)N0 * DD * 2;
constexpr size_t OFF_H1    = OFF_N1BF + (size_t)N1 * DD * 2;
constexpr size_t OFF_H2    = OFF_H1   + (size_t)N1 * DD * 4;
constexpr size_t OFF_HITEM = OFF_H2   + (size_t)N2 * DD * 4;
constexpr size_t OFF_Q0BF  = OFF_HITEM + (size_t)N2 * DD * 4;   // [128][128] transposed
constexpr size_t OFF_W0BF  = OFF_Q0BF + (size_t)DD * DD * 2;    // [128][256] transposed
constexpr size_t OFF_Q1BF  = OFF_W0BF + (size_t)2 * DD * DD * 2;
constexpr size_t OFF_W1BF  = OFF_Q1BF + (size_t)DD * DD * 2;

// ---------------------------------------------------------------------------
// Async global->LDS copy support (CDNA5 GLOBAL_LOAD_ASYNC_TO_LDS_B128).
#if defined(__gfx1250__) && __has_builtin(__builtin_amdgcn_global_load_async_to_lds_b128)
#define HAVE_ASYNC_LDS 1
typedef __attribute__((address_space(1))) v4i as1_v4i;
typedef __attribute__((address_space(3))) v4i as3_v4i;
#define ASYNC_CP16(gp, lp) \
  __builtin_amdgcn_global_load_async_to_lds_b128((as1_v4i*)(gp), (as3_v4i*)(lp), 0, 0)
#else
#define HAVE_ASYNC_LDS 0
#endif

__device__ __forceinline__ void wait_async_all() {
#if HAVE_ASYNC_LDS
#if __has_builtin(__builtin_amdgcn_s_wait_asynccnt)
  __builtin_amdgcn_s_wait_asynccnt(0);
#else
  asm volatile("s_wait_asynccnt 0" ::: "memory");
#endif
#endif
}

// Copy `bytes` (multiple of 4096) from contiguous global `g` to LDS `l`.
// 256 threads, 16B chunks per lane.
__device__ __forceinline__ void stage_lds(const __bf16* g, __bf16* l, int bytes, int t) {
#if HAVE_ASYNC_LDS
  const int chunks = bytes >> 4;
  for (int id = t; id < chunks; id += 256)
    ASYNC_CP16((const char*)g + id * 16, (char*)l + id * 16);
#else
  const int n = bytes >> 1;
  for (int id = t; id < n; id += 256) l[id] = g[id];
#endif
}

// ---------------------------------------------------------------------------
__global__ __launch_bounds__(256) void cvt_transpose_bf16(
    const float* __restrict__ s /*[K][N]*/, __bf16* __restrict__ d /*[N][K]*/,
    int K, int N) {
  int i = blockIdx.x * 256 + threadIdx.x;
  if (i < K * N) { int k = i / N, n = i % N; d[n * K + k] = (__bf16)s[i]; }
}

// --- WMMA fragment builders per CDNA5 ISA 16-bit layouts (wave32) -----------
// A (16x32 bf16): lanes 0-15 hold rows M=0..15 with K = kc*32 + {0..7,16..23};
//                 lanes 16-31 hold same rows with K = kc*32 + {8..15,24..31}.
__device__ __forceinline__ v16bf ldA(const __bf16* Alds, int rowBase, int kc, int lane) {
  const int half = lane >> 4, m = lane & 15;
  const __bf16* row = Alds + (rowBase + m) * DD;
  const int klo = kc * 32 + half * 8;
  const int khi = kc * 32 + 16 + half * 8;
  v16bf a;
#pragma unroll
  for (int e = 0; e < 8; ++e) { a[e] = row[klo + e]; a[8 + e] = row[khi + e]; }
  return a;
}
// B (32x16 bf16), W staged transposed in LDS as Wt[n][k] (ldk = 128):
// lane holds column N = lane&15; elements 0..15 are K = kc*32 + {0..7,16..23} + half*8.
__device__ __forceinline__ v16bf ldB(const __bf16* Wt, int colT, int kc, int lane) {
  const int half = lane >> 4;
  const __bf16* p = Wt + (colT + (lane & 15)) * DD + kc * 32 + half * 8;
  v16bf b;
#pragma unroll
  for (int e = 0; e < 8; ++e) { b[e] = p[e]; b[8 + e] = p[16 + e]; }
  return b;
}

// ---------------------------------------------------------------------------
// out[i,:] = relu(gather(hsrc, idx, i) @ W + bias), stored bf16.
// 64 rows per WG, 256 threads (8 waves); wave (w) does 16 rows x 64 cols.
__global__ __launch_bounds__(256) void gemm_relu_bf16(
    const float* __restrict__ hsrc, const int* __restrict__ idx,
    const __bf16* __restrict__ WT /*[n=128][k=128] transposed*/,
    const float* __restrict__ bias, __bf16* __restrict__ nout) {
  __shared__ __bf16 Wt[DD * DD];   // [n][k], 32 KB
  __shared__ __bf16 A[64 * DD];    // 16 KB
  const int t = threadIdx.x;
  const long blockRow = (long)blockIdx.x * 64;

  stage_lds(WT, Wt, DD * DD * 2, t);             // async weight tile -> LDS
  {                                              // gather + f32->bf16 rows
    int r = t >> 2, c0 = (t & 3) * 32;
    long gr = blockRow + r;
    long sr = idx ? (long)idx[gr] : gr;
    const float* sp = hsrc + sr * DD + c0;
    __bf16* ap = A + r * DD + c0;
#pragma unroll
    for (int j = 0; j < 32; ++j) ap[j] = (__bf16)sp[j];
  }
  wait_async_all();
  __syncthreads();

  const int w = t >> 5, lane = t & 31;
  const int rowBase = (w >> 1) * 16, colBase = (w & 1) * 64;
  const int half = lane >> 4, cl = lane & 15;
  v16bf aF[4];
#pragma unroll
  for (int kc = 0; kc < 4; ++kc) aF[kc] = ldA(A, rowBase, kc, lane);
#pragma unroll
  for (int ct = 0; ct < 4; ++ct) {
    int colT = colBase + ct * 16;
    v8f acc = {0.f, 0.f, 0.f, 0.f, 0.f, 0.f, 0.f, 0.f};
#pragma unroll
    for (int kc = 0; kc < 4; ++kc) {
      v16bf bF = ldB(Wt, colT, kc, lane);
      acc = __builtin_amdgcn_wmma_f32_16x16x32_bf16(false, aF[kc], false, bF,
                                                    (short)0, acc, false, false);
    }
    int col = colT + cl;
    float b = bias[col];
#pragma unroll
    for (int r = 0; r < 8; ++r) {
      long row = blockRow + rowBase + r + half * 8;   // C/D layout: lanes>=16 -> M+8
      nout[row * DD + col] = (__bf16)fmaxf(acc[r] + b, 0.f);
    }
  }
}

// ---------------------------------------------------------------------------
// Segment-sum: macc[dst[e],:] += n[src[e],:] * w[e]; wsum[dst[e]] += w[e].
__global__ __launch_bounds__(256) void scatter_edges(
    const __bf16* __restrict__ nsrc, const int* __restrict__ src,
    const int* __restrict__ dst, const float* __restrict__ w,
    float* __restrict__ macc, float* __restrict__ wsum, int E) {
  int e = blockIdx.x * 8 + (threadIdx.x >> 5);
  if (e >= E) return;
  int lane = threadIdx.x & 31;
  int s = src[e], d = dst[e];
  float wt = w[e];
  int c = lane * 4;
  const __bf16* np = nsrc + (long)s * DD + c;
  float* mp = macc + (long)d * DD + c;
#pragma unroll
  for (int j = 0; j < 4; ++j) atomicAdd(&mp[j], (float)np[j] * wt);
  if (lane == 0) atomicAdd(&wsum[d], wt);
}

// ---------------------------------------------------------------------------
// z = relu([m/ws || h_dst] @ W(256x128) + b); K=256 as two K=128 passes,
// acc kept in registers across passes. WT is [n=128][k=256] transposed.
__global__ __launch_bounds__(256) void zcombine(
    const float* __restrict__ macc, const float* __restrict__ wsum,
    const float* __restrict__ hdst, const int* __restrict__ idx,
    const __bf16* __restrict__ WT, const float* __restrict__ bias,
    float* __restrict__ zout) {
  __shared__ __bf16 Wt[DD * DD];
  __shared__ __bf16 A[64 * DD];
  const int t = threadIdx.x;
  const long blockRow = (long)blockIdx.x * 64;
  const int w = t >> 5, lane = t & 31;
  const int rowBase = (w >> 1) * 16, colBase = (w & 1) * 64;
  const int half = lane >> 4, cl = lane & 15;

  v8f acc[4];
#pragma unroll
  for (int ct = 0; ct < 4; ++ct)
    acc[ct] = (v8f){0.f, 0.f, 0.f, 0.f, 0.f, 0.f, 0.f, 0.f};

  for (int pass = 0; pass < 2; ++pass) {
    // Stage the K-half of W: LDS row n (128 bf16) = WT[n][pass*128 .. +128),
    // each row is 256 contiguous bytes in global.
#if HAVE_ASYNC_LDS
    for (int id = t; id < DD * 16; id += 256) {         // 16 x 16B chunks per row
      int n = id >> 4, c = id & 15;
      ASYNC_CP16(WT + (size_t)n * 256 + pass * DD + c * 8, Wt + n * DD + c * 8);
    }
#else
    for (int id = t; id < DD * DD; id += 256) {
      int n = id >> 7, kl = id & 127;
      Wt[n * DD + kl] = WT[(size_t)n * 256 + pass * DD + kl];
    }
#endif
    {
      int r = t >> 2, c0 = (t & 3) * 32;
      long gr = blockRow + r;
      __bf16* ap = A + r * DD + c0;
      if (pass == 0) {
        float inv = 1.f / fmaxf(wsum[gr], 1.f);   // clip(ws, 1, inf)
        const float* mp = macc + gr * DD + c0;
#pragma unroll
        for (int j = 0; j < 32; ++j) ap[j] = (__bf16)(mp[j] * inv);
      } else {
        long sr = idx ? (long)idx[gr] : gr;
        const float* hp = hdst + sr * DD + c0;
#pragma unroll
        for (int j = 0; j < 32; ++j) ap[j] = (__bf16)hp[j];
      }
    }
    wait_async_all();
    __syncthreads();
    v16bf aF[4];
#pragma unroll
    for (int kc = 0; kc < 4; ++kc) aF[kc] = ldA(A, rowBase, kc, lane);
#pragma unroll
    for (int ct = 0; ct < 4; ++ct) {
      int colT = colBase + ct * 16;
#pragma unroll
      for (int kc = 0; kc < 4; ++kc) {
        v16bf bF = ldB(Wt, colT, kc, lane);
        acc[ct] = __builtin_amdgcn_wmma_f32_16x16x32_bf16(false, aF[kc], false, bF,
                                                          (short)0, acc[ct], false, false);
      }
    }
    __syncthreads();   // before pass 1 overwrites LDS
  }
#pragma unroll
  for (int ct = 0; ct < 4; ++ct) {
    int col = colBase + ct * 16 + cl;
    float b = bias[col];
#pragma unroll
    for (int r = 0; r < 8; ++r) {
      long row = blockRow + rowBase + r + half * 8;
      zout[row * DD + col] = fmaxf(acc[ct][r] + b, 0.f);
    }
  }
}

// ---------------------------------------------------------------------------
__global__ __launch_bounds__(256) void row_normalize(float* __restrict__ h) {
  int row = blockIdx.x * 8 + (threadIdx.x >> 5);
  int lane = threadIdx.x & 31;
  float* p = h + (long)row * DD + lane * 4;
  float4 v = *(const float4*)p;
  float ss = v.x * v.x + v.y * v.y + v.z * v.z + v.w * v.w;
#pragma unroll
  for (int o = 16; o > 0; o >>= 1) ss += __shfl_xor(ss, o, 32);
  float nrm = sqrtf(ss);
  float sc = (nrm > 0.f) ? 1.f / nrm : 1.f;
  float4 r = {v.x * sc, v.y * sc, v.z * sc, v.w * sc};
  *(float4*)p = r;
}

__global__ __launch_bounds__(256) void hitem_kernel(
    const float* __restrict__ emb, const int* __restrict__ nids,
    const float* __restrict__ h2, float* __restrict__ hi) {
  long g = (long)blockIdx.x * 256 + threadIdx.x;
  long row = g >> 7; int c = (int)(g & 127);
  long sr = (long)nids[row];
  hi[g] = emb[sr * DD + c] + h2[g];
}

__global__ __launch_bounds__(256) void score_kernel(
    const float* __restrict__ hi, const float* __restrict__ bias,
    const int* __restrict__ nids,
    const int* __restrict__ ps, const int* __restrict__ pd,
    const int* __restrict__ ns, const int* __restrict__ nd,
    float* __restrict__ out) {
  int p = blockIdx.x * 8 + (threadIdx.x >> 5);
  int lane = threadIdx.x & 31;
  int a = ps[p], b = pd[p], c = ns[p], d = nd[p];
  float4 va = ((const float4*)(hi + (long)a * DD))[lane];
  float4 vb = ((const float4*)(hi + (long)b * DD))[lane];
  float4 vc = ((const float4*)(hi + (long)c * DD))[lane];
  float4 vd = ((const float4*)(hi + (long)d * DD))[lane];
  float sp = va.x * vb.x + va.y * vb.y + va.z * vb.z + va.w * vb.w;
  float sn = vc.x * vd.x + vc.y * vd.y + vc.z * vd.z + vc.w * vd.w;
#pragma unroll
  for (int o = 16; o > 0; o >>= 1) { sp += __shfl_xor(sp, o, 32); sn += __shfl_xor(sn, o, 32); }
  if (lane == 0) {
    float bp = bias[nids[a]] + bias[nids[b]];
    float bn = bias[nids[c]] + bias[nids[d]];
    out[p] = fmaxf((sn + bn) - (sp + bp) + 1.f, 0.f);
  }
}

// ---------------------------------------------------------------------------
extern "C" void kernel_launch(void* const* d_in, const int* in_sizes, int n_in,
                              void* d_out, int out_size, void* d_ws, size_t ws_size,
                              hipStream_t stream) {
  (void)in_sizes; (void)n_in; (void)out_size; (void)ws_size;
  const float* emb  = (const float*)d_in[0];
  const float* bias = (const float*)d_in[1];
  const int*   nids = (const int*)d_in[2];
  const int*   src0 = (const int*)d_in[3];
  const int*   dst0 = (const int*)d_in[4];
  const float* w0   = (const float*)d_in[5];
  const int*   src1 = (const int*)d_in[6];
  const int*   dst1 = (const int*)d_in[7];
  const float* w1   = (const float*)d_in[8];
  const int*   pos_src = (const int*)d_in[9];
  const int*   pos_dst = (const int*)d_in[10];
  const int*   neg_src = (const int*)d_in[11];
  const int*   neg_dst = (const int*)d_in[12];
  const float* Q0w = (const float*)d_in[13];
  const float* Q0b = (const float*)d_in[14];
  const float* W0w = (const float*)d_in[15];
  const float* W0b = (const float*)d_in[16];
  const float* Q1w = (const float*)d_in[17];
  const float* Q1b = (const float*)d_in[18];
  const float* W1w = (const float*)d_in[19];
  const float* W1b = (const float*)d_in[20];

  char* ws = (char*)d_ws;
  float*  macc0 = (float*)(ws + OFF_MACC0);
  float*  wsum0 = (float*)(ws + OFF_WSUM0);
  float*  macc1 = (float*)(ws + OFF_MACC1);
  float*  wsum1 = (float*)(ws + OFF_WSUM1);
  __bf16* n0bf  = (__bf16*)(ws + OFF_N0BF);
  __bf16* n1bf  = (__bf16*)(ws + OFF_N1BF);
  float*  h1    = (float*)(ws + OFF_H1);
  float*  h2    = (float*)(ws + OFF_H2);
  float*  hitem = (float*)(ws + OFF_HITEM);
  __bf16* q0bf  = (__bf16*)(ws + OFF_Q0BF);
  __bf16* w0bf  = (__bf16*)(ws + OFF_W0BF);
  __bf16* q1bf  = (__bf16*)(ws + OFF_Q1BF);
  __bf16* w1bf  = (__bf16*)(ws + OFF_W1BF);

  (void)hipMemsetAsync(d_ws, 0, ZERO_BYTES, stream);  // zero segment-sum accumulators

  // bf16 + transpose weights once: QT [128][128], WT [128][256]
  cvt_transpose_bf16<<<(DD * DD + 255) / 256, 256, 0, stream>>>(Q0w, q0bf, DD, DD);
  cvt_transpose_bf16<<<(2 * DD * DD + 255) / 256, 256, 0, stream>>>(W0w, w0bf, 2 * DD, DD);
  cvt_transpose_bf16<<<(DD * DD + 255) / 256, 256, 0, stream>>>(Q1w, q1bf, DD, DD);
  cvt_transpose_bf16<<<(2 * DD * DD + 255) / 256, 256, 0, stream>>>(W1w, w1bf, 2 * DD, DD);

  // Layer 1
  gemm_relu_bf16<<<N0 / 64, 256, 0, stream>>>(emb, nids, q0bf, Q0b, n0bf);
  scatter_edges<<<E0 / 8, 256, 0, stream>>>(n0bf, src0, dst0, w0, macc0, wsum0, E0);
  zcombine<<<N1 / 64, 256, 0, stream>>>(macc0, wsum0, emb, nids, w0bf, W0b, h1);
  row_normalize<<<N1 / 8, 256, 0, stream>>>(h1);

  // Layer 2
  gemm_relu_bf16<<<N1 / 64, 256, 0, stream>>>(h1, nullptr, q1bf, Q1b, n1bf);
  scatter_edges<<<E1 / 8, 256, 0, stream>>>(n1bf, src1, dst1, w1, macc1, wsum1, E1);
  zcombine<<<N2 / 64, 256, 0, stream>>>(macc1, wsum1, h1, nullptr, w1bf, W1b, h2);
  row_normalize<<<N2 / 8, 256, 0, stream>>>(h2);

  // Scoring
  hitem_kernel<<<N2 * DD / 256, 256, 0, stream>>>(emb, nids, h2, hitem);
  score_kernel<<<EP / 8, 256, 0, stream>>>(hitem, bias, nids,
                                           pos_src, pos_dst, neg_src, neg_dst,
                                           (float*)d_out);
}
